// HANModel_23708219474081
// MI455X (gfx1250) — compile-verified, matrix operand order checked by reading, not run on hardware
//
#include <hip/hip_runtime.h>
#include <hip/hip_bf16.h>
#include <math.h>

// ---------------- types ----------------
typedef _Float16 v16h __attribute__((ext_vector_type(16)));
typedef _Float16 v8h  __attribute__((ext_vector_type(8)));
typedef float    v8f  __attribute__((ext_vector_type(8)));

// ---------------- problem constants ----------------
#define S_   256
#define W_   64
#define E_   256
#define HW_  256
#define HS_  256
#define C_   10
#define G3   768         // 3*H
#define MROWS 16384      // S*W

// ---------------- workspace layout (bytes) ----------------
#define OFF_WIHF  ((size_t)0)
#define OFF_WIHB  ((size_t)393216)
#define OFF_WHHF  ((size_t)786432)
#define OFF_WHHB  ((size_t)1179648)
#define OFF_SIHF  ((size_t)1572864)
#define OFF_SIHB  ((size_t)2359296)
#define OFF_SHHF  ((size_t)3145728)
#define OFF_SHHB  ((size_t)3538944)
#define OFF_XG    ((size_t)3932160)    // 2*16384*768 f16
#define OFF_WOUT  ((size_t)54263808)   // 16384*512 f16
#define OFF_SV    ((size_t)71041024)   // 256*512 f32
#define OFF_SVH   ((size_t)71565312)   // 256*512 f16
#define OFF_SXG   ((size_t)71827456)   // 2*256*768 f16
#define OFF_SOUT  ((size_t)72613888)   // 256*512 f16

// ---------------- WMMA helpers ----------------
__device__ __forceinline__ v8f wmma16(v16h a, v16h b, v8f c) {
  return __builtin_amdgcn_wmma_f32_16x16x32_f16(false, a, false, b, (short)0, c,
                                                false, false);
}

// A fragment (16x32 f16) from LDS tile stored row-major [16][ldk]
// lane 0..15 -> M=lane, K = kt*32 + {0..7, 16..23}
// lane 16..31-> M=lane-16, K = kt*32 + {8..15, 24..31}
__device__ __forceinline__ v16h load_a_frag(const _Float16* At, int ldk,
                                            int lane, int kt) {
  int m  = lane & 15;
  int hi = lane >> 4;
  const _Float16* p = At + m * ldk + kt * 32 + 8 * hi;
  v8h p0 = *(const v8h*)(p);        // ds_load_b128
  v8h p1 = *(const v8h*)(p + 16);   // ds_load_b128
  return __builtin_shufflevector(p0, p1, 0, 1, 2, 3, 4, 5, 6, 7,
                                 8, 9, 10, 11, 12, 13, 14, 15);
}

// B fragment: weights pre-swizzled so each lane reads 16 contiguous halves
__device__ __forceinline__ v16h load_b_frag(const _Float16* Bf, int KT,
                                            int nt, int kt, int lane) {
  return *(const v16h*)(Bf + ((((size_t)nt * KT + kt) * 32 + lane) << 4));
}

__device__ __forceinline__ float sigm(float x) {
  return 1.0f / (1.0f + __expf(-x));
}

// ---------------- kernel 0: weight -> f16 B-fragment swizzle ----------------
// src: W [N rows][K cols] row-major f32.  B[k][n] = W[n][k].
// dst element idx = ((nt*KT + kt)*32 + lane)*16 + j
__global__ void prep_frag_kernel(const float* __restrict__ src,
                                 _Float16* __restrict__ dst, int N, int K) {
  int idx = blockIdx.x * blockDim.x + threadIdx.x;
  int total = N * K;
  if (idx >= total) return;
  int KT   = K >> 5;
  int j    = idx & 15;
  int lane = (idx >> 4) & 31;
  int f    = idx >> 9;
  int kt   = f % KT;
  int nt   = f / KT;
  int n    = nt * 16 + (lane & 15);
  int hi   = lane >> 4;
  int koff = (j < 8) ? (8 * hi + j) : (16 + 8 * hi + (j - 8));
  int k    = kt * 32 + koff;
  dst[idx] = (_Float16)src[(size_t)n * K + k];
}

// ---------------- kernel 1: word input gates (embedding gather + GEMM) -----
// XG[dir][r][g] = (emb[tokens[r]] . Wih[g,:]) + bih[g]   (f16 out)
__global__ void word_xg_kernel(const int* __restrict__ tokens,
                               const float* __restrict__ emb,
                               const _Float16* __restrict__ fragF,
                               const _Float16* __restrict__ fragB,
                               const float* __restrict__ bihF,
                               const float* __restrict__ bihB,
                               _Float16* __restrict__ XG) {
  __shared__ _Float16 At[16][E_];
  const int r0  = blockIdx.x * 16;
  const int dir = blockIdx.y;
  const int tid = threadIdx.x;

  // gather 16 embedding rows -> LDS f16 (coalesced)
  for (int rep = 0; rep < 16; ++rep) {
    int tok = tokens[r0 + rep];
    At[rep][tid] = (_Float16)emb[(size_t)tok * E_ + tid];
  }
  __syncthreads();

  const _Float16* Bf  = dir ? fragB : fragF;
  const float*    bih = dir ? bihB : bihF;
  const int wave = tid >> 5, lane = tid & 31;

  v16h a[8];
#pragma unroll
  for (int kt = 0; kt < 8; ++kt) a[kt] = load_a_frag(&At[0][0], E_, lane, kt);

  const int n  = lane & 15;
  const int hi = lane >> 4;
  const size_t base = ((size_t)dir * MROWS + r0) * G3;

#pragma unroll
  for (int i = 0; i < 6; ++i) {
    int nt = wave * 6 + i;
    v8f acc = {0.f, 0.f, 0.f, 0.f, 0.f, 0.f, 0.f, 0.f};
#pragma unroll
    for (int kt = 0; kt < 8; ++kt)
      acc = wmma16(a[kt], load_b_frag(Bf, 8, nt, kt, lane), acc);
    int g = nt * 16 + n;
    float bb = bih[g];
#pragma unroll
    for (int v = 0; v < 8; ++v) {
      int m = v + 8 * hi;
      XG[base + (size_t)m * G3 + g] = (_Float16)(acc[v] + bb);
    }
  }
}

// ---------------- kernel 2: word-level GRU recurrence ----------------------
// One WG owns 16 sentences + one direction; loops 64 timesteps internally.
__global__ void word_rec_kernel(const _Float16* __restrict__ XG,
                                const _Float16* __restrict__ fragF,
                                const _Float16* __restrict__ fragB,
                                const float* __restrict__ bhhF,
                                const float* __restrict__ bhhB,
                                _Float16* __restrict__ wordOut) {
  __shared__ _Float16 hH[16][HW_];
  __shared__ float    hF[16][HW_];
  __shared__ float    HG[16][G3];
  const int s0  = blockIdx.x * 16;
  const int dir = blockIdx.y;
  const int tid = threadIdx.x;
  const int wave = tid >> 5, lane = tid & 31;
  const _Float16* Bf  = dir ? fragB : fragF;
  const float*    bhh = dir ? bhhB : bhhF;

  for (int i = 0; i < 16; ++i) {
    int idx = i * 256 + tid;
    hH[idx >> 8][idx & 255] = (_Float16)0.f;
    hF[idx >> 8][idx & 255] = 0.f;
  }
  __syncthreads();

  const int nlo = lane & 15, hi = lane >> 4;

  for (int step = 0; step < W_; ++step) {
    const int t = dir ? (W_ - 1 - step) : step;

    // HG = hH @ Whh^T
    v16h a[8];
#pragma unroll
    for (int kt = 0; kt < 8; ++kt) a[kt] = load_a_frag(&hH[0][0], HW_, lane, kt);
#pragma unroll
    for (int i = 0; i < 6; ++i) {
      int nt = wave * 6 + i;
      v8f acc = {0.f, 0.f, 0.f, 0.f, 0.f, 0.f, 0.f, 0.f};
#pragma unroll
      for (int kt = 0; kt < 8; ++kt)
        acc = wmma16(a[kt], load_b_frag(Bf, 8, nt, kt, lane), acc);
#pragma unroll
      for (int v = 0; v < 8; ++v) HG[v + 8 * hi][nt * 16 + nlo] = acc[v];
    }
    __syncthreads();

    // fused GRU gates
    for (int i = 0; i < 16; ++i) {
      int idx = i * 256 + tid;
      int m = idx >> 8, j = idx & 255;
      size_t r = (size_t)(s0 + m) * W_ + t;
      const _Float16* xg = XG + ((size_t)dir * MROWS + r) * G3;
      float xr = (float)xg[j], xz = (float)xg[HW_ + j], xn = (float)xg[2 * HW_ + j];
      float hr = HG[m][j] + bhh[j];
      float hz = HG[m][HW_ + j] + bhh[HW_ + j];
      float hn = HG[m][2 * HW_ + j] + bhh[2 * HW_ + j];
      float rr = sigm(xr + hr);
      float zz = sigm(xz + hz);
      float nn = tanhf(xn + rr * hn);
      float hnew = (1.f - zz) * nn + zz * hF[m][j];
      hF[m][j] = hnew;
      hH[m][j] = (_Float16)hnew;
      wordOut[r * (2 * HW_) + dir * HW_ + j] = (_Float16)hnew;
    }
    __syncthreads();
  }
}

// ---------------- kernel 3: word attention pooling -------------------------
__global__ void word_attn_kernel(const _Float16* __restrict__ wordOut,
                                 const float* __restrict__ attW,
                                 const float* __restrict__ attB,
                                 float* __restrict__ sentVec,
                                 _Float16* __restrict__ sentVecH) {
  __shared__ float sc[W_];
  const int s = blockIdx.x, tid = threadIdx.x;
  if (tid < W_) {
    float sum = 0.f;
    const _Float16* row = wordOut + ((size_t)s * W_ + tid) * (2 * HW_);
    for (int d = 0; d < 2 * HW_; ++d) sum += (float)row[d] * attW[d];
    sc[tid] = tanhf(sum + attB[0]);
  }
  __syncthreads();
  if (tid == 0) {
    float mx = -1e30f;
    for (int t = 0; t < W_; ++t) mx = fmaxf(mx, sc[t]);
    float se = 0.f;
    for (int t = 0; t < W_; ++t) { float e = __expf(sc[t] - mx); sc[t] = e; se += e; }
    float inv = 1.f / se;
    for (int t = 0; t < W_; ++t) sc[t] *= inv;
  }
  __syncthreads();
  float a0 = 0.f, a1 = 0.f;
  for (int t = 0; t < W_; ++t) {
    float al = sc[t];
    const _Float16* row = wordOut + ((size_t)s * W_ + t) * (2 * HW_);
    a0 += al * (float)row[tid];
    a1 += al * (float)row[tid + 256];
  }
  sentVec[(size_t)s * 512 + tid]        = a0;
  sentVec[(size_t)s * 512 + tid + 256]  = a1;
  sentVecH[(size_t)s * 512 + tid]       = (_Float16)a0;
  sentVecH[(size_t)s * 512 + tid + 256] = (_Float16)a1;
}

// ---------------- kernel 4: sentence input gates (K=512 GEMM) --------------
__global__ void sent_xg_kernel(const _Float16* __restrict__ svH,
                               const _Float16* __restrict__ fragF,
                               const _Float16* __restrict__ fragB,
                               const float* __restrict__ bihF,
                               const float* __restrict__ bihB,
                               _Float16* __restrict__ SXG) {
  __shared__ _Float16 At[16][512];
  const int s0  = blockIdx.x * 16;
  const int dir = blockIdx.y;
  const int tid = threadIdx.x;
  for (int i = 0; i < 32; ++i) {
    int idx = i * 256 + tid;
    At[idx >> 9][idx & 511] = svH[(size_t)(s0 + (idx >> 9)) * 512 + (idx & 511)];
  }
  __syncthreads();

  const _Float16* Bf  = dir ? fragB : fragF;
  const float*    bih = dir ? bihB : bihF;
  const int wave = tid >> 5, lane = tid & 31;
  const int n = lane & 15, hi = lane >> 4;

#pragma unroll
  for (int i = 0; i < 6; ++i) {
    int nt = wave * 6 + i;
    v8f acc = {0.f, 0.f, 0.f, 0.f, 0.f, 0.f, 0.f, 0.f};
#pragma unroll
    for (int kt = 0; kt < 16; ++kt) {
      v16h a = load_a_frag(&At[0][0], 512, lane, kt);
      acc = wmma16(a, load_b_frag(Bf, 16, nt, kt, lane), acc);
    }
    int g = nt * 16 + n;
    float bb = bih[g];
#pragma unroll
    for (int v = 0; v < 8; ++v) {
      int m = v + 8 * hi;
      SXG[((size_t)dir * S_ + s0 + m) * G3 + g] = (_Float16)(acc[v] + bb);
    }
  }
}

// ---------------- kernel 5: sentence-level GRU (batch 1, M padded to 16) ---
__global__ void sent_rec_kernel(const _Float16* __restrict__ SXG,
                                const _Float16* __restrict__ fragF,
                                const _Float16* __restrict__ fragB,
                                const float* __restrict__ bhhF,
                                const float* __restrict__ bhhB,
                                _Float16* __restrict__ sentOut) {
  __shared__ _Float16 hH[16][HS_];
  __shared__ float    hF[HS_];
  __shared__ float    HG0[G3];
  const int dir = blockIdx.x;
  const int tid = threadIdx.x;
  const int wave = tid >> 5, lane = tid & 31;
  const _Float16* Bf  = dir ? fragB : fragF;
  const float*    bhh = dir ? bhhB : bhhF;

  for (int i = 0; i < 16; ++i) {
    int idx = i * 256 + tid;
    hH[idx >> 8][idx & 255] = (_Float16)0.f;
  }
  hF[tid] = 0.f;
  __syncthreads();

  const int nlo = lane & 15, hi = lane >> 4;

  for (int step = 0; step < S_; ++step) {
    const int t = dir ? (S_ - 1 - step) : step;
    v16h a[8];
#pragma unroll
    for (int kt = 0; kt < 8; ++kt) a[kt] = load_a_frag(&hH[0][0], HS_, lane, kt);
#pragma unroll
    for (int i = 0; i < 6; ++i) {
      int nt = wave * 6 + i;
      v8f acc = {0.f, 0.f, 0.f, 0.f, 0.f, 0.f, 0.f, 0.f};
#pragma unroll
      for (int kt = 0; kt < 8; ++kt)
        acc = wmma16(a[kt], load_b_frag(Bf, 8, nt, kt, lane), acc);
      if (hi == 0) HG0[nt * 16 + nlo] = acc[0];  // only row M=0 is real
    }
    __syncthreads();

    {
      int j = tid;
      const _Float16* xg = SXG + ((size_t)dir * S_ + t) * G3;
      float xr = (float)xg[j], xz = (float)xg[HS_ + j], xn = (float)xg[2 * HS_ + j];
      float hr = HG0[j] + bhh[j];
      float hz = HG0[HS_ + j] + bhh[HS_ + j];
      float hn = HG0[2 * HS_ + j] + bhh[2 * HS_ + j];
      float rr = sigm(xr + hr);
      float zz = sigm(xz + hz);
      float nn = tanhf(xn + rr * hn);
      float hnew = (1.f - zz) * nn + zz * hF[j];
      hF[j] = hnew;
      hH[0][j] = (_Float16)hnew;
      sentOut[(size_t)t * 512 + dir * HS_ + j] = (_Float16)hnew;
    }
    __syncthreads();
  }
}

// ---------------- kernel 6: sentence attention + classifier + softmax ------
__global__ void final_kernel(const _Float16* __restrict__ sentOut,
                             const float* __restrict__ attW,
                             const float* __restrict__ attB,
                             const float* __restrict__ clsW,
                             const float* __restrict__ clsB,
                             float* __restrict__ out) {
  __shared__ float sc[S_];
  __shared__ float dv[512];
  __shared__ float lg[C_];
  const int tid = threadIdx.x;

  {
    float sum = 0.f;
    const _Float16* row = sentOut + (size_t)tid * 512;
    for (int d = 0; d < 512; ++d) sum += (float)row[d] * attW[d];
    sc[tid] = tanhf(sum + attB[0]);
  }
  __syncthreads();
  if (tid == 0) {
    float mx = -1e30f;
    for (int s = 0; s < S_; ++s) mx = fmaxf(mx, sc[s]);
    float se = 0.f;
    for (int s = 0; s < S_; ++s) { float e = __expf(sc[s] - mx); sc[s] = e; se += e; }
    float inv = 1.f / se;
    for (int s = 0; s < S_; ++s) sc[s] *= inv;
  }
  __syncthreads();
  {
    float a0 = 0.f, a1 = 0.f;
    for (int s = 0; s < S_; ++s) {
      float al = sc[s];
      a0 += al * (float)sentOut[(size_t)s * 512 + tid];
      a1 += al * (float)sentOut[(size_t)s * 512 + tid + 256];
    }
    dv[tid] = a0;
    dv[tid + 256] = a1;
  }
  __syncthreads();
  if (tid < C_) {
    float sum = clsB[tid];
    for (int d = 0; d < 512; ++d) sum += dv[d] * clsW[tid * 512 + d];
    lg[tid] = sum;
  }
  __syncthreads();
  if (tid == 0) {
    float mx = -1e30f;
    for (int c = 0; c < C_; ++c) mx = fmaxf(mx, lg[c]);
    float se = 0.f;
    float e[C_];
    for (int c = 0; c < C_; ++c) { e[c] = __expf(lg[c] - mx); se += e[c]; }
    float inv = 1.f / se;
    for (int c = 0; c < C_; ++c) out[c] = e[c] * inv;
  }
}

// ---------------- launch ----------------
extern "C" void kernel_launch(void* const* d_in, const int* in_sizes, int n_in,
                              void* d_out, int out_size, void* d_ws, size_t ws_size,
                              hipStream_t stream) {
  (void)in_sizes; (void)n_in; (void)out_size; (void)ws_size;
  const int*   tokens = (const int*)d_in[0];
  const float* emb    = (const float*)d_in[1];
  const float* wWihF  = (const float*)d_in[2];
  const float* wWhhF  = (const float*)d_in[3];
  const float* wBihF  = (const float*)d_in[4];
  const float* wBhhF  = (const float*)d_in[5];
  const float* wWihB  = (const float*)d_in[6];
  const float* wWhhB  = (const float*)d_in[7];
  const float* wBihB  = (const float*)d_in[8];
  const float* wBhhB  = (const float*)d_in[9];
  const float* sWihF  = (const float*)d_in[10];
  const float* sWhhF  = (const float*)d_in[11];
  const float* sBihF  = (const float*)d_in[12];
  const float* sBhhF  = (const float*)d_in[13];
  const float* sWihB  = (const float*)d_in[14];
  const float* sWhhB  = (const float*)d_in[15];
  const float* sBihB  = (const float*)d_in[16];
  const float* sBhhB  = (const float*)d_in[17];
  const float* wAttW  = (const float*)d_in[18];
  const float* wAttB  = (const float*)d_in[19];
  const float* sAttW  = (const float*)d_in[20];
  const float* sAttB  = (const float*)d_in[21];
  const float* clsW   = (const float*)d_in[22];
  const float* clsB   = (const float*)d_in[23];

  char* ws = (char*)d_ws;
  _Float16* wihF = (_Float16*)(ws + OFF_WIHF);
  _Float16* wihB = (_Float16*)(ws + OFF_WIHB);
  _Float16* whhF = (_Float16*)(ws + OFF_WHHF);
  _Float16* whhB = (_Float16*)(ws + OFF_WHHB);
  _Float16* sihF = (_Float16*)(ws + OFF_SIHF);
  _Float16* sihB = (_Float16*)(ws + OFF_SIHB);
  _Float16* shhF = (_Float16*)(ws + OFF_SHHF);
  _Float16* shhB = (_Float16*)(ws + OFF_SHHB);
  _Float16* XG   = (_Float16*)(ws + OFF_XG);
  _Float16* wOut = (_Float16*)(ws + OFF_WOUT);
  float*    sVec = (float*)   (ws + OFF_SV);
  _Float16* sVecH= (_Float16*)(ws + OFF_SVH);
  _Float16* SXG  = (_Float16*)(ws + OFF_SXG);
  _Float16* sOut = (_Float16*)(ws + OFF_SOUT);

  // weight swizzle to WMMA fragment order (f16)
  prep_frag_kernel<<<768, 256, 0, stream>>>(wWihF, wihF, G3, E_);
  prep_frag_kernel<<<768, 256, 0, stream>>>(wWihB, wihB, G3, E_);
  prep_frag_kernel<<<768, 256, 0, stream>>>(wWhhF, whhF, G3, HW_);
  prep_frag_kernel<<<768, 256, 0, stream>>>(wWhhB, whhB, G3, HW_);
  prep_frag_kernel<<<1536, 256, 0, stream>>>(sWihF, sihF, G3, 512);
  prep_frag_kernel<<<1536, 256, 0, stream>>>(sWihB, sihB, G3, 512);
  prep_frag_kernel<<<768, 256, 0, stream>>>(sWhhF, shhF, G3, HS_);
  prep_frag_kernel<<<768, 256, 0, stream>>>(sWhhB, shhB, G3, HS_);

  // word level
  word_xg_kernel<<<dim3(MROWS / 16, 2), 256, 0, stream>>>(
      tokens, emb, wihF, wihB, wBihF, wBihB, XG);
  word_rec_kernel<<<dim3(S_ / 16, 2), 256, 0, stream>>>(
      XG, whhF, whhB, wBhhF, wBhhB, wOut);
  word_attn_kernel<<<S_, 256, 0, stream>>>(wOut, wAttW, wAttB, sVec, sVecH);

  // sentence level
  sent_xg_kernel<<<dim3(S_ / 16, 2), 256, 0, stream>>>(
      sVecH, sihF, sihB, sBihF, sBihB, SXG);
  sent_rec_kernel<<<2, 256, 0, stream>>>(SXG, shhF, shhB, sBhhF, sBhhB, sOut);

  // attention + classifier + softmax
  final_kernel<<<1, 256, 0, stream>>>(sOut, sAttW, sAttB, clsW, clsB,
                                      (float*)d_out);
}
